// QuietSTaRModelWrapper_61675730370660
// MI455X (gfx1250) — matrix operand order repaired and checked by reading, not compiled, not run on hardware
//
#include <hip/hip_runtime.h>

// ---------------------------------------------------------------------------
// QuietSTaR forward for MI455X (gfx1250): all GEMMs via v_wmma_f32_16x16x32_bf16
// ---------------------------------------------------------------------------

typedef __bf16 bf16_t;
typedef __attribute__((ext_vector_type(8)))  __bf16 v8bf;
typedef __attribute__((ext_vector_type(16))) __bf16 v16bf;
typedef __attribute__((ext_vector_type(8)))  float  v8f;

#define CDIV(a, b) (((a) + (b) - 1) / (b))

__device__ __forceinline__ bf16_t f2bf(float f) {
  unsigned u = __builtin_bit_cast(unsigned, f);
  u += 0x7fffu + ((u >> 16) & 1u);           // round-to-nearest-even
  unsigned short s = (unsigned short)(u >> 16);
  return __builtin_bit_cast(bf16_t, s);
}

__device__ __forceinline__ v8f vzero8() {
  v8f v;
#pragma unroll
  for (int i = 0; i < 8; ++i) v[i] = 0.0f;
  return v;
}

__device__ __forceinline__ v8f wmma_bf16(v16bf a, v16bf b, v8f c) {
  // D = A(16x32 bf16) * B(32x16 bf16) + C(16x16 f32)
  return __builtin_amdgcn_wmma_f32_16x16x32_bf16(false, a, false, b,
                                                 (short)0, c, false, false);
}

// A-fragment (16x32, MxK): lane holds row m=lane&15; half-wave selects K-chunk.
// VGPR0-3: K = kh*8 .. kh*8+7 ; VGPR4-7: K = 16+kh*8 .. (kh = lane>>4)
__device__ __forceinline__ v16bf load_a_frag(const bf16_t* A, int lda, int m0,
                                             int k, int lane) {
  int mm = m0 + (lane & 15);
  int kh = (lane >> 4) * 8;
  const bf16_t* base = A + (size_t)mm * lda + k + kh;
  v8bf lo = *(const v8bf*)(base);
  v8bf hi = *(const v8bf*)(base + 16);
  v16bf r;
#pragma unroll
  for (int i = 0; i < 8; ++i) { r[i] = lo[i]; r[8 + i] = hi[i]; }
  return r;
}

// B-fragment (32x16, KxN) from row-major W[N,K]: lane holds column n=lane&15;
// VGPR0-3: K = kh*8 .. kh*8+7 ; VGPR4-7: K = 16+kh*8 .. (kh = lane>>4)
__device__ __forceinline__ v16bf load_b_frag(const bf16_t* W, int ldw, int n0,
                                             int k, int lane) {
  int nn = n0 + (lane & 15);
  int kh = (lane >> 4) * 8;
  const bf16_t* base = W + (size_t)nn * ldw + k + kh;
  v8bf lo = *(const v8bf*)(base);
  v8bf hi = *(const v8bf*)(base + 16);
  v16bf r;
#pragma unroll
  for (int i = 0; i < 8; ++i) { r[i] = lo[i]; r[8 + i] = hi[i]; }
  return r;
}

// ---------------------------------------------------------------------------
// Generic tiled WMMA GEMM: C[M,N] = A[M,K] * W[N,K]^T (+epilogue)
// Block: 256 threads (8 waves); block tile 128x128; wave tile 32(M) x 64(N):
// each B-fragment feeds two WMMAs (B reuse across the 2 row sub-tiles).
// ---------------------------------------------------------------------------
struct GP {
  const bf16_t* A;  int lda;
  const bf16_t* W;             // [N,K] row-major, ld = K
  const float*  bias;          // [N] or null
  const float*  res;           // residual [M,N] f32 or null
  float*        outF;
  bf16_t*       outB;
  int M, N, K;
  const int* sot; const int* eot;
};

// EPI: 1=relu->bf16   2=f32(+bias+res)   3=Q/K head-major bf16
//      4=V transposed bf16   5=f32(+bias)   6=logits f32 (-inf @ sot/eot)
template <int EPI>
__global__ __launch_bounds__(256) void gemm_kernel(GP p) {
  const int lane = threadIdx.x & 31;
  const int wave = threadIdx.x >> 5;
  const int wm = wave & 3, wn = wave >> 2;
  const int row0 = blockIdx.y * 128 + wm * 32;
  const int col0 = blockIdx.x * 128 + wn * 64;

  v8f c[2][4];
#pragma unroll
  for (int a = 0; a < 2; ++a)
#pragma unroll
    for (int f = 0; f < 4; ++f) c[a][f] = vzero8();

  for (int k = 0; k < p.K; k += 32) {
    if (k + 32 < p.K)
      __builtin_prefetch(p.A + (size_t)(row0 + (lane & 15)) * p.lda + k + 32, 0, 0);
    v16bf a0 = load_a_frag(p.A, p.lda, row0,      k, lane);
    v16bf a1 = load_a_frag(p.A, p.lda, row0 + 16, k, lane);
#pragma unroll
    for (int f = 0; f < 4; ++f) {
      v16bf b = load_b_frag(p.W, p.K, col0 + f * 16, k, lane);
      c[0][f] = wmma_bf16(a0, b, c[0][f]);
      c[1][f] = wmma_bf16(a1, b, c[1][f]);
    }
  }

  const int nbase = col0 + (lane & 15);
  const int mbase = row0 + ((lane >> 4) << 3);
  int sot = 0, eot = 0;
  if (EPI == 6) { sot = *p.sot; eot = *p.eot; }

#pragma unroll
  for (int a = 0; a < 2; ++a) {
#pragma unroll
    for (int f = 0; f < 4; ++f) {
      const int n = nbase + f * 16;
      const float bia = p.bias ? p.bias[n] : 0.0f;
#pragma unroll
      for (int r = 0; r < 8; ++r) {
        const int m = mbase + a * 16 + r;
        float v = c[a][f][r] + bia;
        if (EPI == 1) {
          p.outB[(size_t)m * p.N + n] = f2bf(v > 0.0f ? v : 0.0f);
        } else if (EPI == 2) {
          p.outF[(size_t)m * p.N + n] = v + p.res[(size_t)m * p.N + n];
        } else if (EPI == 3) {        // q/k: [b,h,s,d]
          int b = m >> 11, s = m & 2047, h = n >> 7, d = n & 127;
          p.outB[((size_t)(b * 8 + h) * 2048 + s) * 128 + d] = f2bf(v);
        } else if (EPI == 4) {        // v^T: [b,h,d,s]
          int b = m >> 11, s = m & 2047, h = n >> 7, d = n & 127;
          p.outB[((size_t)(b * 8 + h) * 128 + d) * 2048 + s] = f2bf(v);
        } else if (EPI == 5) {
          p.outF[(size_t)m * p.N + n] = v;
        } else if (EPI == 6) {
          p.outF[(size_t)m * p.N + n] =
              (n == sot || n == eot) ? -__builtin_inff() : v;
        }
      }
    }
  }
}

// ---------------------------------------------------------------------------
// Flash-style attention. Keys restricted by thought_mask (mask on keys only).
// Block: 128 threads (4 waves); block tile = 64 queries, wave = 16 queries.
// grid = (B*NH, S/64)
// ---------------------------------------------------------------------------
__global__ __launch_bounds__(128) void attn_kernel(const bf16_t* __restrict__ qg,
                                                   const bf16_t* __restrict__ kg,
                                                   const bf16_t* __restrict__ vt,
                                                   const unsigned char* __restrict__ mask,
                                                   bf16_t* __restrict__ ctx) {
  constexpr int S = 2048, HD = 128, H = 1024;
  const int lane = threadIdx.x & 31;
  const int wave = threadIdx.x >> 5;
  const int bh = blockIdx.x;                  // 0..15
  const int b = bh >> 3, h = bh & 7;
  const int q0 = blockIdx.y * 64 + wave * 16;

  __shared__ __align__(32) bf16_t pshare[4][16 * 32];
  bf16_t* myP = pshare[wave];

  const bf16_t* qbase = qg + (size_t)bh * S * HD;
  const bf16_t* kbase = kg + (size_t)bh * S * HD;
  const bf16_t* vbase = vt + (size_t)bh * HD * S;
  const unsigned char* mk = mask + (size_t)b * S;

  v16bf qa[4];
#pragma unroll
  for (int j = 0; j < 4; ++j) qa[j] = load_a_frag(qbase, HD, q0, j * 32, lane);

  v8f acc[8];
#pragma unroll
  for (int f = 0; f < 8; ++f) acc[f] = vzero8();
  float rm[8], rs[8];
#pragma unroll
  for (int r = 0; r < 8; ++r) { rm[r] = -__builtin_inff(); rs[r] = 0.0f; }

  const float scale = 0.0883883476483184f;    // 1/sqrt(128)

  for (int kt = 0; kt < S; kt += 32) {
    v8f s0 = vzero8(), s1 = vzero8();
#pragma unroll
    for (int j = 0; j < 4; ++j) {
      v16bf b0 = load_b_frag(kbase, HD, kt,      j * 32, lane);
      v16bf b1 = load_b_frag(kbase, HD, kt + 16, j * 32, lane);
      s0 = wmma_bf16(qa[j], b0, s0);
      s1 = wmma_bf16(qa[j], b1, s1);
    }
    const bool m0v = mk[kt + (lane & 15)] != 0;
    const bool m1v = mk[kt + 16 + (lane & 15)] != 0;

    float sc0[8], sc1[8];
#pragma unroll
    for (int r = 0; r < 8; ++r) {
      sc0[r] = m0v ? s0[r] * scale : -1e9f;
      sc1[r] = m1v ? s1[r] * scale : -1e9f;
    }
    // Online softmax per query row (rows live in fixed 16-lane halves).
#pragma unroll
    for (int r = 0; r < 8; ++r) {
      float tm = fmaxf(sc0[r], sc1[r]);
#pragma unroll
      for (int x = 8; x > 0; x >>= 1) tm = fmaxf(tm, __shfl_xor(tm, x, 32));
      const float nm = fmaxf(rm[r], tm);
      const float al = __expf(rm[r] - nm);    // 0 on first tile (rm = -inf)
      const float p0 = __expf(sc0[r] - nm);
      const float p1 = __expf(sc1[r] - nm);
      float ts = p0 + p1;
#pragma unroll
      for (int x = 8; x > 0; x >>= 1) ts += __shfl_xor(ts, x, 32);
      rs[r] = rs[r] * al + ts;
      rm[r] = nm;
      sc0[r] = p0; sc1[r] = p1;
#pragma unroll
      for (int f = 0; f < 8; ++f) acc[f][r] *= al;
    }
    // P (C-layout) -> LDS -> P (A-fragment layout)
    {
      const int col = lane & 15;
      const int rb = (lane >> 4) << 3;
#pragma unroll
      for (int r = 0; r < 8; ++r) {
        myP[(rb + r) * 32 + col]      = f2bf(sc0[r]);
        myP[(rb + r) * 32 + 16 + col] = f2bf(sc1[r]);
      }
    }
    asm volatile("s_wait_dscnt 0x0" ::: "memory");
    v16bf pa;
    {
      const int m = lane & 15;
      const int kh = (lane >> 4) * 8;
      v8bf lo = *(const v8bf*)(myP + m * 32 + kh);
      v8bf hi = *(const v8bf*)(myP + m * 32 + 16 + kh);
#pragma unroll
      for (int i = 0; i < 8; ++i) { pa[i] = lo[i]; pa[8 + i] = hi[i]; }
    }
    // ctx += P(16x32) * V(32x16 per d-group); vt rows are d (ld = S)
#pragma unroll
    for (int f = 0; f < 8; ++f) {
      v16bf vb = load_b_frag(vbase, S, f * 16, kt, lane);
      acc[f] = wmma_bf16(pa, vb, acc[f]);
    }
  }

  const int n = lane & 15;
  const int mb = (lane >> 4) << 3;
#pragma unroll
  for (int f = 0; f < 8; ++f) {
#pragma unroll
    for (int r = 0; r < 8; ++r) {
      const int q = q0 + mb + r;
      ctx[((size_t)b * S + q) * H + h * 128 + f * 16 + n] = f2bf(acc[f][r] / rs[r]);
    }
  }
}

// ---------------------------------------------------------------------------
// Elementwise / reduction kernels
// ---------------------------------------------------------------------------
__device__ __forceinline__ float blockSum256(float v, float* sred) {
#pragma unroll
  for (int x = 16; x > 0; x >>= 1) v += __shfl_xor(v, x, 32);
  const int w = threadIdx.x >> 5;
  if ((threadIdx.x & 31) == 0) sred[w] = v;
  __syncthreads();
  float tot = 0.0f;
#pragma unroll
  for (int i = 0; i < 8; ++i) tot += sred[i];
  __syncthreads();
  return tot;
}

__global__ __launch_bounds__(256) void cvt_kernel(const float* __restrict__ src,
                                                  bf16_t* __restrict__ dst,
                                                  size_t n, int srcRow, int dstRow) {
  size_t i = (size_t)blockIdx.x * 256 + threadIdx.x;
  if (i >= n) return;
  size_t r = i / (size_t)srcRow, c = i % (size_t)srcRow;
  dst[r * dstRow + c] = f2bf(src[i]);
}

__global__ __launch_bounds__(256) void ln1_kernel(const float* __restrict__ y,
                                                  const float* __restrict__ g,
                                                  const float* __restrict__ bt,
                                                  float* __restrict__ xf,
                                                  bf16_t* __restrict__ xb) {
  __shared__ float sred[8];
  const int row = blockIdx.x;
  const float* yr = y + (size_t)row * 1024;
  float v[4], s = 0.0f;
#pragma unroll
  for (int i = 0; i < 4; ++i) { v[i] = yr[threadIdx.x + i * 256]; s += v[i]; }
  const float mean = blockSum256(s, sred) * (1.0f / 1024.0f);
  float s2 = 0.0f;
#pragma unroll
  for (int i = 0; i < 4; ++i) { float d = v[i] - mean; s2 += d * d; }
  const float var = blockSum256(s2, sred) * (1.0f / 1024.0f);
  const float rstd = rsqrtf(var + 1e-5f);
#pragma unroll
  for (int i = 0; i < 4; ++i) {
    const int c = threadIdx.x + i * 256;
    const float o = (v[i] - mean) * rstd * g[c] + bt[c];
    xf[(size_t)row * 1024 + c] = o;
    xb[(size_t)row * 1024 + c] = f2bf(o);
  }
}

// LN2 + scatter-select (thought mask) + gate dot-product, te -> cat[:,1024:]
__global__ __launch_bounds__(256) void ln2_kernel(const float* __restrict__ y,
                                                  const float* __restrict__ g,
                                                  const float* __restrict__ bt,
                                                  const float* __restrict__ hs,
                                                  const unsigned char* __restrict__ mask,
                                                  const float* __restrict__ gate_w,
                                                  const float* __restrict__ gate_b,
                                                  bf16_t* __restrict__ cat,
                                                  float* __restrict__ gate) {
  __shared__ float sred[8];
  const int row = blockIdx.x;
  const bool is_t = mask[row] != 0;
  const float* yr = y + (size_t)row * 1024;
  float v[4], s = 0.0f;
#pragma unroll
  for (int i = 0; i < 4; ++i) { v[i] = yr[threadIdx.x + i * 256]; s += v[i]; }
  const float mean = blockSum256(s, sred) * (1.0f / 1024.0f);
  float s2 = 0.0f;
#pragma unroll
  for (int i = 0; i < 4; ++i) { float d = v[i] - mean; s2 += d * d; }
  const float var = blockSum256(s2, sred) * (1.0f / 1024.0f);
  const float rstd = rsqrtf(var + 1e-5f);
  float gdot = 0.0f;
#pragma unroll
  for (int i = 0; i < 4; ++i) {
    const int c = threadIdx.x + i * 256;
    const float enc = (v[i] - mean) * rstd * g[c] + bt[c];
    const float te = is_t ? enc : hs[(size_t)row * 1024 + c];
    cat[(size_t)row * 2048 + 1024 + c] = f2bf(te);
    gdot += te * gate_w[c];
  }
  const float tot = blockSum256(gdot, sred);
  if (threadIdx.x == 0)
    gate[row] = 1.0f / (1.0f + __expf(-(tot + gate_b[0])));
}

__global__ __launch_bounds__(256) void mix_kernel(const float* __restrict__ hs,
                                                  const float* __restrict__ mc,
                                                  const float* __restrict__ gate,
                                                  const unsigned char* __restrict__ mask,
                                                  bf16_t* __restrict__ out) {
  const size_t i = (size_t)blockIdx.x * 256 + threadIdx.x;  // exact multiple
  const int row = (int)(i >> 10);
  const float eff = (mask[row] != 0) ? gate[row] : 0.1f;
  out[i] = f2bf((1.0f - eff) * hs[i] + eff * mc[i]);
}

// ---------------------------------------------------------------------------
// Host launcher
// ---------------------------------------------------------------------------
extern "C" void kernel_launch(void* const* d_in, const int* in_sizes, int n_in,
                              void* d_out, int out_size, void* d_ws, size_t ws_size,
                              hipStream_t stream) {
  (void)in_sizes; (void)n_in; (void)out_size; (void)ws_size;
  constexpr int B = 2, S = 2048, H = 1024, NH = 8, HD = 128, FF = 4096, V = 32000;
  constexpr int M = B * S;                     // 4096

  const float* hs   = (const float*)d_in[0];
  const unsigned char* mask = (const unsigned char*)d_in[1];
  const int* sot    = (const int*)d_in[2];
  const int* eot    = (const int*)d_in[3];
  const float* wq   = (const float*)d_in[4];
  const float* bq   = (const float*)d_in[5];
  const float* wk   = (const float*)d_in[6];
  const float* bk   = (const float*)d_in[7];
  const float* wv   = (const float*)d_in[8];
  const float* bv   = (const float*)d_in[9];
  const float* wo   = (const float*)d_in[10];
  const float* bo   = (const float*)d_in[11];
  const float* ln1g = (const float*)d_in[12];
  const float* ln1b = (const float*)d_in[13];
  const float* wff1 = (const float*)d_in[14];
  const float* bff1 = (const float*)d_in[15];
  const float* wff2 = (const float*)d_in[16];
  const float* bff2 = (const float*)d_in[17];
  const float* ln2g = (const float*)d_in[18];
  const float* ln2b = (const float*)d_in[19];
  const float* gatew= (const float*)d_in[20];
  const float* gateb= (const float*)d_in[21];
  const float* mixw = (const float*)d_in[22];
  const float* mixb = (const float*)d_in[23];
  const float* outw = (const float*)d_in[24];
  const float* outb = (const float*)d_in[25];

  size_t off = 0;
  auto alloc = [&](size_t bytes) -> char* {
    char* p = (char*)d_ws + off;
    off = (off + bytes + 255) & ~(size_t)255;
    return p;
  };
  bf16_t* wqb   = (bf16_t*)alloc((size_t)H * H * 2);
  bf16_t* wkb   = (bf16_t*)alloc((size_t)H * H * 2);
  bf16_t* wvb   = (bf16_t*)alloc((size_t)H * H * 2);
  bf16_t* wob   = (bf16_t*)alloc((size_t)H * H * 2);
  bf16_t* wff1b = (bf16_t*)alloc((size_t)FF * H * 2);
  bf16_t* wff2b = (bf16_t*)alloc((size_t)H * FF * 2);
  bf16_t* mixwb = (bf16_t*)alloc((size_t)H * 2 * H * 2);
  bf16_t* outwb = (bf16_t*)alloc((size_t)V * H * 2);
  bf16_t* cat   = (bf16_t*)alloc((size_t)M * 2 * H * 2);   // [hs_bf | te_bf]
  bf16_t* qb    = (bf16_t*)alloc((size_t)M * H * 2);
  bf16_t* kb    = (bf16_t*)alloc((size_t)M * H * 2);
  bf16_t* vtb   = (bf16_t*)alloc((size_t)M * H * 2);
  bf16_t* ctxb  = (bf16_t*)alloc((size_t)M * H * 2);
  float*  yf    = (float*)alloc((size_t)M * H * 4);        // reused scratch
  float*  x1f   = (float*)alloc((size_t)M * H * 4);
  bf16_t* x1b   = (bf16_t*)alloc((size_t)M * H * 2);
  bf16_t* h1b   = (bf16_t*)alloc((size_t)M * FF * 2);
  float*  gatef = (float*)alloc((size_t)M * 4);
  bf16_t* mixdb = (bf16_t*)alloc((size_t)M * H * 2);

  auto cvt = [&](const float* s, bf16_t* d, size_t n, int sr, int dr) {
    cvt_kernel<<<(unsigned)CDIV(n, 256), 256, 0, stream>>>(s, d, n, sr, dr);
  };
  // 1) fp32 -> bf16 conversions (weights + hs into cat first half)
  cvt(wq,   wqb,   (size_t)H * H,     H,  H);
  cvt(wk,   wkb,   (size_t)H * H,     H,  H);
  cvt(wv,   wvb,   (size_t)H * H,     H,  H);
  cvt(wo,   wob,   (size_t)H * H,     H,  H);
  cvt(wff1, wff1b, (size_t)FF * H,    H,  H);
  cvt(wff2, wff2b, (size_t)H * FF,    FF, FF);
  cvt(mixw, mixwb, (size_t)H * 2 * H, 2 * H, 2 * H);
  cvt(outw, outwb, (size_t)V * H,     H,  H);
  cvt(hs,   cat,   (size_t)M * H,     H,  2 * H);

  GP p{};
  const dim3 blk(256);
  // 2) QKV projections (head-major layouts; V transposed)
  p = {cat, 2 * H, wqb, bq, nullptr, nullptr, qb,  M, H, H, nullptr, nullptr};
  gemm_kernel<3><<<dim3(H / 128, M / 128), blk, 0, stream>>>(p);
  p = {cat, 2 * H, wkb, bk, nullptr, nullptr, kb,  M, H, H, nullptr, nullptr};
  gemm_kernel<3><<<dim3(H / 128, M / 128), blk, 0, stream>>>(p);
  p = {cat, 2 * H, wvb, bv, nullptr, nullptr, vtb, M, H, H, nullptr, nullptr};
  gemm_kernel<4><<<dim3(H / 128, M / 128), blk, 0, stream>>>(p);

  // 3) masked attention -> ctx
  attn_kernel<<<dim3(B * NH, S / 64), dim3(128), 0, stream>>>(qb, kb, vtb, mask, ctxb);

  // 4) out-proj + residual -> yf ; LN1 -> x1
  p = {ctxb, H, wob, bo, hs, yf, nullptr, M, H, H, nullptr, nullptr};
  gemm_kernel<2><<<dim3(H / 128, M / 128), blk, 0, stream>>>(p);
  ln1_kernel<<<M, blk, 0, stream>>>(yf, ln1g, ln1b, x1f, x1b);

  // 5) FFN
  p = {x1b, H, wff1b, bff1, nullptr, nullptr, h1b, M, FF, H, nullptr, nullptr};
  gemm_kernel<1><<<dim3(FF / 128, M / 128), blk, 0, stream>>>(p);
  p = {h1b, FF, wff2b, bff2, x1f, yf, nullptr, M, H, FF, nullptr, nullptr};
  gemm_kernel<2><<<dim3(H / 128, M / 128), blk, 0, stream>>>(p);

  // 6) LN2 + mask-select + gate; te -> cat[:,1024:]
  ln2_kernel<<<M, blk, 0, stream>>>(yf, ln2g, ln2b, hs, mask, gatew, gateb, cat, gatef);

  // 7) mixer: concat(hs, te) @ mixer_w^T + b -> yf ; then blend
  p = {cat, 2 * H, mixwb, mixb, nullptr, yf, nullptr, M, H, 2 * H, nullptr, nullptr};
  gemm_kernel<5><<<dim3(H / 128, M / 128), blk, 0, stream>>>(p);
  mix_kernel<<<(M * H) / 256, blk, 0, stream>>>(hs, yf, gatef, mask, mixdb);

  // 8) vocab projection -> logits (f32) with -inf at sot/eot
  p = {mixdb, H, outwb, outb, nullptr, (float*)d_out, nullptr, M, V, H, sot, eot};
  gemm_kernel<6><<<dim3(V / 128, M / 128), blk, 0, stream>>>(p);
}
